// GAT_11209864642900
// MI455X (gfx1250) — matrix operand order repaired and checked by reading, not compiled, not run on hardware
//
#include <hip/hip_runtime.h>
#include <hip/hip_bf16.h>
#include <cstdint>

typedef __attribute__((ext_vector_type(16))) _Float16 v16h;
typedef __attribute__((ext_vector_type(8)))  _Float16 v8h;
typedef __attribute__((ext_vector_type(8)))  float    v8f;

#define NEG_SLOPE 0.2f

// ---- monotone float <-> uint encoding for atomicMax-based segment max ----
__device__ __forceinline__ unsigned enc_f32(float f) {
  unsigned u = __float_as_uint(f);
  return (u & 0x80000000u) ? ~u : (u | 0x80000000u);
}
__device__ __forceinline__ float dec_f32(unsigned u) {
  unsigned b = (u & 0x80000000u) ? (u ^ 0x80000000u) : ~u;
  return __uint_as_float(b);
}
// enc==0 decodes to NaN -> "no incoming edges", matching the reference's
// where(isfinite(m), m, 0).

__device__ __forceinline__ void edge_nodes(int e, int E, const int* src, const int* dst,
                                           int& s, int& d) {
  if (e < E) { s = src[e]; d = dst[e]; } else { s = e - E; d = s; }
}

__device__ __forceinline__ v16h cvt_a_frag(float4 f0, float4 f1, float4 f2, float4 f3) {
  v16h a;
  a[0]=(_Float16)f0.x;  a[1]=(_Float16)f0.y;  a[2]=(_Float16)f0.z;  a[3]=(_Float16)f0.w;
  a[4]=(_Float16)f1.x;  a[5]=(_Float16)f1.y;  a[6]=(_Float16)f1.z;  a[7]=(_Float16)f1.w;
  a[8]=(_Float16)f2.x;  a[9]=(_Float16)f2.y;  a[10]=(_Float16)f2.z; a[11]=(_Float16)f2.w;
  a[12]=(_Float16)f3.x; a[13]=(_Float16)f3.y; a[14]=(_Float16)f3.z; a[15]=(_Float16)f3.w;
  return a;
}

#define WMMA_F16(A_, B_, C_) \
  __builtin_amdgcn_wmma_f32_16x16x32_f16(false, (A_), false, (B_), (short)0, (C_), false, false)

// =====================  WMMA GEMM: D = A(f32,MxK) * Bt^T  ==================
// Bt is the f16 TRANSPOSED weight: Bt[n*K + k] = W[k*N + n].
// One wave computes a 32x64 output tile: 2 A fragments x 4 B fragments =
// 8 v_wmma_f32_16x16x32_f16 per K-step.  A stream double-buffered (HBM/L2),
// B single-buffered (weights are L0/L2 resident).
template<bool OUT_F16>
__global__ __launch_bounds__(256)
void k_gemm_wmma(const float* __restrict__ A, const _Float16* __restrict__ Bt,
                 void* __restrict__ Dout, int M, int K, int Nout)
{
  const int lane = threadIdx.x & 31;
  const int wib  = threadIdx.x >> 5;
  const int wpb  = blockDim.x >> 5;
  const long wid = (long)blockIdx.x * wpb + wib;
  const int ntg  = Nout >> 6;                      // 64-column groups
  const long mtiles2 = (long)((M >> 4) + 1) >> 1;  // 32-row tiles
  const long mt  = wid / ntg;
  const int  ng  = (int)(wid % ntg);
  if (mt >= mtiles2) return;

  const int l15  = lane & 15;
  const int kbA  = (lane < 16) ? 0 : 8;            // A: interleaved 8-chunks (ISA 7.12.2)
  const int kbB  = (lane < 16) ? 0 : 8;            // B: interleaved 8-chunks (same as A)
  const int ncol = (ng << 6) + l15;

  const int  mbase = (int)(mt << 5);
  const bool hiOk  = (mbase + 31) < M;             // last tile may be a half tile
  const int  rowLo = mbase + l15;
  const int  rowHi = hiOk ? (mbase + 16 + l15) : rowLo;  // clamp -> valid loads
  const float* aLo = A + (size_t)rowLo * K + kbA;
  const float* aHi = A + (size_t)rowHi * K + kbA;
  const _Float16* bq0 = Bt + (size_t)(ncol)      * K + kbB;
  const _Float16* bq1 = Bt + (size_t)(ncol + 16) * K + kbB;
  const _Float16* bq2 = Bt + (size_t)(ncol + 32) * K + kbB;
  const _Float16* bq3 = Bt + (size_t)(ncol + 48) * K + kbB;

  v8f cL0 = {}, cL1 = {}, cL2 = {}, cL3 = {};
  v8f cH0 = {}, cH1 = {}, cH2 = {}, cH3 = {};

  // prime the A pipeline (k0 = 0)
  float4 l0 = *(const float4*)(aLo);      float4 l1 = *(const float4*)(aLo + 4);
  float4 l2 = *(const float4*)(aLo + 16); float4 l3 = *(const float4*)(aLo + 20);
  float4 h0 = *(const float4*)(aHi);      float4 h1 = *(const float4*)(aHi + 4);
  float4 h2 = *(const float4*)(aHi + 16); float4 h3 = *(const float4*)(aHi + 20);

  for (int k0 = 0; k0 < K; k0 += 32) {
    // B for this K-step (short-latency: L0/L2 resident weights)
    v16h b0 = __builtin_shufflevector(*(const v8h*)(bq0 + k0), *(const v8h*)(bq0 + k0 + 16),
                                      0,1,2,3,4,5,6,7,8,9,10,11,12,13,14,15);
    v16h b1 = __builtin_shufflevector(*(const v8h*)(bq1 + k0), *(const v8h*)(bq1 + k0 + 16),
                                      0,1,2,3,4,5,6,7,8,9,10,11,12,13,14,15);
    v16h b2 = __builtin_shufflevector(*(const v8h*)(bq2 + k0), *(const v8h*)(bq2 + k0 + 16),
                                      0,1,2,3,4,5,6,7,8,9,10,11,12,13,14,15);
    v16h b3 = __builtin_shufflevector(*(const v8h*)(bq3 + k0), *(const v8h*)(bq3 + k0 + 16),
                                      0,1,2,3,4,5,6,7,8,9,10,11,12,13,14,15);

    // start next iteration's A stream loads (one full K-step ahead)
    float4 nl0, nl1, nl2, nl3, nh0, nh1, nh2, nh3;
    const int kn = k0 + 32;
    const bool more = kn < K;
    if (more) {
      const float* pl = aLo + kn;
      const float* ph = aHi + kn;
      nl0 = *(const float4*)(pl);      nl1 = *(const float4*)(pl + 4);
      nl2 = *(const float4*)(pl + 16); nl3 = *(const float4*)(pl + 20);
      nh0 = *(const float4*)(ph);      nh1 = *(const float4*)(ph + 4);
      nh2 = *(const float4*)(ph + 16); nh3 = *(const float4*)(ph + 20);
      __builtin_prefetch(pl + 64, 0, 3);   // WGP-scope prefetch, 2 K-steps ahead
      __builtin_prefetch(ph + 64, 0, 3);
    }

    // convert current A and run the 8-WMMA chain
    v16h alo = cvt_a_frag(l0, l1, l2, l3);
    v16h ahi = cvt_a_frag(h0, h1, h2, h3);
    cL0 = WMMA_F16(alo, b0, cL0);
    cH0 = WMMA_F16(ahi, b0, cH0);
    cL1 = WMMA_F16(alo, b1, cL1);
    cH1 = WMMA_F16(ahi, b1, cH1);
    cL2 = WMMA_F16(alo, b2, cL2);
    cH2 = WMMA_F16(ahi, b2, cH2);
    cL3 = WMMA_F16(alo, b3, cL3);
    cH3 = WMMA_F16(ahi, b3, cH3);

    if (more) {
      l0 = nl0; l1 = nl1; l2 = nl2; l3 = nl3;
      h0 = nh0; h1 = nh1; h2 = nh2; h3 = nh3;
    }
  }

  // C/D layout: VGPR r -> M = r + 8*laneHalf, N = lane%16 within the 16-col tile
  const int mOffLane = (lane < 16) ? 0 : 8;
  if constexpr (OUT_F16) {
    _Float16* D = (_Float16*)Dout;
    for (int r = 0; r < 8; ++r) {
      size_t base = (size_t)(mbase + mOffLane + r) * Nout + ncol;
      D[base]      = (_Float16)cL0[r];
      D[base + 16] = (_Float16)cL1[r];
      D[base + 32] = (_Float16)cL2[r];
      D[base + 48] = (_Float16)cL3[r];
    }
    if (hiOk) {
      for (int r = 0; r < 8; ++r) {
        size_t base = (size_t)(mbase + 16 + mOffLane + r) * Nout + ncol;
        D[base]      = (_Float16)cH0[r];
        D[base + 16] = (_Float16)cH1[r];
        D[base + 32] = (_Float16)cH2[r];
        D[base + 48] = (_Float16)cH3[r];
      }
    }
  } else {
    float* D = (float*)Dout;
    for (int r = 0; r < 8; ++r) {
      size_t base = (size_t)(mbase + mOffLane + r) * Nout + ncol;
      D[base]      = cL0[r];
      D[base + 16] = cL1[r];
      D[base + 32] = cL2[r];
      D[base + 48] = cL3[r];
    }
    if (hiOk) {
      for (int r = 0; r < 8; ++r) {
        size_t base = (size_t)(mbase + 16 + mOffLane + r) * Nout + ncol;
        D[base]      = cH0[r];
        D[base + 16] = cH1[r];
        D[base + 32] = cH2[r];
        D[base + 48] = cH3[r];
      }
    }
  }
}

// ============================ helper kernels ===============================
__global__ void k_fill_u32(unsigned* p, long n, unsigned v) {
  long t = (long)blockIdx.x * blockDim.x + threadIdx.x;
  if (t < n) p[t] = v;
}

__global__ void k_wt_f16(const float* __restrict__ W, _Float16* __restrict__ Wt,
                         int K, int Nout) {
  long t = (long)blockIdx.x * blockDim.x + threadIdx.x;
  if (t >= (long)K * Nout) return;
  int k = (int)(t / Nout), n = (int)(t % Nout);
  Wt[(size_t)n * K + k] = (_Float16)W[t];
}

// per-(node,head) attention logits a_src, a_dst
template<typename FT>
__global__ void k_att(const FT* __restrict__ feat, const float* __restrict__ atts,
                      const float* __restrict__ attd, float* __restrict__ as_,
                      float* __restrict__ ad_, int N, int H, int hshift, int C) {
  long t = (long)blockIdx.x * blockDim.x + threadIdx.x;
  if (t >= (long)N * H) return;
  int n = (int)(t >> hshift), h = (int)(t & (H - 1));
  const FT* f = feat + (size_t)n * H * C + (size_t)h * C;
  float s = 0.f, d = 0.f;
  for (int c = 0; c < C; ++c) {
    float v = (float)f[c];
    s += v * atts[h * C + c];
    d += v * attd[h * C + c];
  }
  as_[t] = s; ad_[t] = d;
}

__global__ void k_edge_max(const int* __restrict__ src, const int* __restrict__ dst,
                           int E, int Etot, const float* __restrict__ as_,
                           const float* __restrict__ ad_, unsigned* __restrict__ mx,
                           int H, int hshift) {
  long t = (long)blockIdx.x * blockDim.x + threadIdx.x;
  if (t >= (long)Etot * H) return;
  int e = (int)(t >> hshift), h = (int)(t & (H - 1));
  int s, d; edge_nodes(e, E, src, dst, s, d);
  float l = as_[s * H + h] + ad_[d * H + h];
  l = l > 0.f ? l : NEG_SLOPE * l;
  atomicMax(&mx[d * H + h], enc_f32(l));
}

__global__ void k_edge_sum(const int* __restrict__ src, const int* __restrict__ dst,
                           int E, int Etot, const float* __restrict__ as_,
                           const float* __restrict__ ad_, const unsigned* __restrict__ mx,
                           float* __restrict__ sm, int H, int hshift) {
  long t = (long)blockIdx.x * blockDim.x + threadIdx.x;
  if (t >= (long)Etot * H) return;
  int e = (int)(t >> hshift), h = (int)(t & (H - 1));
  int s, d; edge_nodes(e, E, src, dst, s, d);
  float l = as_[s * H + h] + ad_[d * H + h];
  l = l > 0.f ? l : NEG_SLOPE * l;
  float m = dec_f32(mx[d * H + h]);
  if (!__builtin_isfinite(m)) m = 0.f;
  atomicAdd(&sm[d * H + h], expf(l - m));
}

// one wave32 per edge: scatter alpha * feat[src] into acc[dst]
template<typename FT>
__global__ void k_edge_aggr(const int* __restrict__ src, const int* __restrict__ dst,
                            int E, int Etot, const float* __restrict__ as_,
                            const float* __restrict__ ad_, const unsigned* __restrict__ mx,
                            const float* __restrict__ sm, const FT* __restrict__ feat,
                            float* __restrict__ acc, int H, int cshift) {
  int lane = threadIdx.x & 31;
  long wv = ((long)blockIdx.x * blockDim.x + threadIdx.x) >> 5;
  if (wv >= Etot) return;
  int e = (int)wv;
  int s, d; edge_nodes(e, E, src, dst, s, d);
  int h = lane & (H - 1);
  float l = as_[s * H + h] + ad_[d * H + h];
  l = l > 0.f ? l : NEG_SLOPE * l;
  float m = dec_f32(mx[d * H + h]);
  if (!__builtin_isfinite(m)) m = 0.f;
  float alpha = expf(l - m) / (sm[d * H + h] + 1e-16f);
  const int HCl = H << cshift;
  const FT* fs = feat + (size_t)s * HCl;
  float* ac = acc + (size_t)d * HCl;
  for (int c = lane; c < HCl; c += 32) {
    float al = __shfl(alpha, c >> cshift, 32);
    atomicAdd(&ac[c], al * (float)fs[c]);
  }
}

__global__ void k_bias_elu(float* __restrict__ acc, const float* __restrict__ b,
                           long total, int HC) {
  long t = (long)blockIdx.x * blockDim.x + threadIdx.x;
  if (t >= total) return;
  float v = acc[t] + b[(int)(t % HC)];
  acc[t] = v > 0.f ? v : expm1f(v);
}

// one wave32 per node, C == 64: Z = zacc + b2; out = log_softmax(Z); write both
__global__ void k_logsoftmax(const float* __restrict__ zacc, const float* __restrict__ b2,
                             float* __restrict__ out, int N, int C) {
  int lane = threadIdx.x & 31;
  long wv = ((long)blockIdx.x * blockDim.x + threadIdx.x) >> 5;
  if (wv >= N) return;
  size_t base = (size_t)wv * C;
  float z0 = zacc[base + lane] + b2[lane];
  float z1 = zacc[base + 32 + lane] + b2[32 + lane];
  float mx = fmaxf(z0, z1);
  for (int o = 16; o > 0; o >>= 1) mx = fmaxf(mx, __shfl_xor(mx, o, 32));
  float se = expf(z0 - mx) + expf(z1 - mx);
  for (int o = 16; o > 0; o >>= 1) se += __shfl_xor(se, o, 32);
  float lse = mx + logf(se);
  out[base + lane] = z0 - lse;
  out[base + 32 + lane] = z1 - lse;
  float* Zout = out + (size_t)N * C;
  Zout[base + lane] = z0;
  Zout[base + 32 + lane] = z1;
}

// ================================ driver ===================================
extern "C" void kernel_launch(void* const* d_in, const int* in_sizes, int n_in,
                              void* d_out, int out_size, void* d_ws, size_t ws_size,
                              hipStream_t stream) {
  const float* x      = (const float*)d_in[0];
  const int*   ei     = (const int*)d_in[1];
  const float* W1     = (const float*)d_in[2];
  const float* att_s1 = (const float*)d_in[3];
  const float* att_d1 = (const float*)d_in[4];
  const float* b1     = (const float*)d_in[5];
  const float* W2     = (const float*)d_in[6];
  const float* att_s2 = (const float*)d_in[7];
  const float* att_d2 = (const float*)d_in[8];
  const float* b2     = (const float*)d_in[9];

  const int HC   = in_sizes[5];            // 512 = heads*hid
  const int FI   = in_sizes[2] / HC;       // 512
  const int N    = in_sizes[0] / FI;       // 50000 (multiple of 16)
  const int E    = in_sizes[1] / 2;        // 400000
  const int CLS  = in_sizes[9];            // 64
  const int H    = 8;                      // heads (reference constant)
  const int HID  = HC / H;                 // 64
  const int Etot = E + N;                  // edges + self loops
  const int* srcp = ei;
  const int* dstp = ei + E;

  // workspace carve-out (~187 MB)
  char* w = (char*)d_ws;
  auto alloc = [&](size_t bytes) { char* p = w; w += (bytes + 255) & ~(size_t)255; return p; };
  _Float16* w1t  = (_Float16*)alloc((size_t)HC * FI * 2);
  _Float16* w2t  = (_Float16*)alloc((size_t)CLS * HC * 2);
  _Float16* h1h  = (_Float16*)alloc((size_t)N * HC * 2);    // layer-1 features (f16)
  float*    as1  = (float*)alloc((size_t)N * H * 4);
  float*    ad1  = (float*)alloc((size_t)N * H * 4);
  unsigned* mx1  = (unsigned*)alloc((size_t)N * H * 4);
  float*    sm1  = (float*)alloc((size_t)N * H * 4);
  float*    acc1 = (float*)alloc((size_t)N * HC * 4);       // aggregated layer-1 out
  float*    z0   = (float*)alloc((size_t)N * CLS * 4);      // layer-2 features (f32)
  float*    as2  = (float*)alloc((size_t)N * 4);
  float*    ad2  = (float*)alloc((size_t)N * 4);
  unsigned* mx2  = (unsigned*)alloc((size_t)N * 4);
  float*    sm2  = (float*)alloc((size_t)N * 4);
  float*    zacc = (float*)alloc((size_t)N * CLS * 4);

  const int TB = 256;
  auto cdiv = [](long a, long b) { return (int)((a + b - 1) / b); };

  // weight transposes -> f16
  k_wt_f16<<<cdiv((long)FI * HC, TB), TB, 0, stream>>>(W1, w1t, FI, HC);
  k_wt_f16<<<cdiv((long)HC * CLS, TB), TB, 0, stream>>>(W2, w2t, HC, CLS);

  // zero/init accumulators (enc 0 == "empty segment" for max)
  k_fill_u32<<<cdiv((long)N * H, TB), TB, 0, stream>>>(mx1, (long)N * H, 0u);
  k_fill_u32<<<cdiv((long)N * H, TB), TB, 0, stream>>>((unsigned*)sm1, (long)N * H, 0u);
  k_fill_u32<<<cdiv((long)N * HC, TB), TB, 0, stream>>>((unsigned*)acc1, (long)N * HC, 0u);
  k_fill_u32<<<cdiv((long)N, TB), TB, 0, stream>>>(mx2, (long)N, 0u);
  k_fill_u32<<<cdiv((long)N, TB), TB, 0, stream>>>((unsigned*)sm2, (long)N, 0u);
  k_fill_u32<<<cdiv((long)N * CLS, TB), TB, 0, stream>>>((unsigned*)zacc, (long)N * CLS, 0u);

  const long mt2 = (long)((N / 16) + 1) / 2;   // 32-row WMMA tiles

  // ---- layer 1 ----
  long wv1 = mt2 * (HC / 64);
  k_gemm_wmma<true><<<cdiv(wv1 * 32, TB), TB, 0, stream>>>(x, w1t, h1h, N, FI, HC);
  k_att<_Float16><<<cdiv((long)N * H, TB), TB, 0, stream>>>(h1h, att_s1, att_d1, as1, ad1, N, H, 3, HID);
  k_edge_max<<<cdiv((long)Etot * H, TB), TB, 0, stream>>>(srcp, dstp, E, Etot, as1, ad1, mx1, H, 3);
  k_edge_sum<<<cdiv((long)Etot * H, TB), TB, 0, stream>>>(srcp, dstp, E, Etot, as1, ad1, mx1, sm1, H, 3);
  k_edge_aggr<_Float16><<<cdiv((long)Etot * 32, TB), TB, 0, stream>>>(
      srcp, dstp, E, Etot, as1, ad1, mx1, sm1, h1h, acc1, H, 6);
  k_bias_elu<<<cdiv((long)N * HC, TB), TB, 0, stream>>>(acc1, b1, (long)N * HC, HC);

  // ---- layer 2 ----
  long wv2 = mt2 * (CLS / 64);
  k_gemm_wmma<false><<<cdiv(wv2 * 32, TB), TB, 0, stream>>>(acc1, w2t, z0, N, HC, CLS);
  k_att<float><<<cdiv((long)N, TB), TB, 0, stream>>>(z0, att_s2, att_d2, as2, ad2, N, 1, 0, CLS);
  k_edge_max<<<cdiv((long)Etot, TB), TB, 0, stream>>>(srcp, dstp, E, Etot, as2, ad2, mx2, 1, 0);
  k_edge_sum<<<cdiv((long)Etot, TB), TB, 0, stream>>>(srcp, dstp, E, Etot, as2, ad2, mx2, sm2, 1, 0);
  k_edge_aggr<float><<<cdiv((long)Etot * 32, TB), TB, 0, stream>>>(
      srcp, dstp, E, Etot, as2, ad2, mx2, sm2, z0, zacc, 1, 6);

  // bias + log_softmax + write (out, Z)
  k_logsoftmax<<<cdiv((long)N * 32, TB), TB, 0, stream>>>(zacc, b2, (float*)d_out, N, CLS);
}
